// AttentionRankingSerializer_55860344652205
// MI455X (gfx1250) — matrix operand order repaired and checked by reading, not compiled, not run on hardware
//
#include <hip/hip_runtime.h>
#include <math.h>

typedef __attribute__((ext_vector_type(16))) __bf16 v16bf;
typedef __attribute__((ext_vector_type(8)))  float  v8f;

#define LEAK 0.2f
#define NSPLIT 4            // column-split factor for the attention kernel

union AB { v16bf v; unsigned int u[8]; };

static __device__ __forceinline__ unsigned short f2bf(float f) {
  unsigned int u = __float_as_uint(f);
  u += 0x7FFFu + ((u >> 16) & 1u);     // round-to-nearest-even
  return (unsigned short)(u >> 16);
}

static __device__ __forceinline__ void atomicMaxF(float* a, float v) {
  int old = __float_as_int(*a);
  while (__int_as_float(old) < v) {
    int assumed = old;
    old = atomicCAS((int*)a, assumed, __float_as_int(v));
    if (old == assumed) break;
  }
}

// ---------- elementwise helpers ----------
__global__ void k_cvt_bf16(const float* __restrict__ src, unsigned short* __restrict__ dst,
                           float scale, int n) {
  int i = blockIdx.x * blockDim.x + threadIdx.x;
  if (i < n) dst[i] = f2bf(src[i] * scale);
}

__global__ void k_fill(float* p, float v, int n) {
  int i = blockIdx.x * blockDim.x + threadIdx.x;
  if (i < n) p[i] = v;
}

// ---------- GEMM: Y[m][n] = sum_k A[m][k] * W[n][k]  (A: MxK bf16, W: NxK bf16, row-major)
// One wave per 16x16 tile, v_wmma_f32_16x16x32_bf16 along K.
__global__ void k_gemm_bf16(const unsigned short* __restrict__ A,
                            const unsigned short* __restrict__ W,
                            float* __restrict__ outF, unsigned short* __restrict__ outB,
                            float scale, int M, int N, int K) {
  int wave = (blockIdx.x * blockDim.x + threadIdx.x) >> 5;
  int lane = threadIdx.x & 31;
  int ntile = N >> 4;
  int nw = (M >> 4) * ntile;
  if (wave >= nw) return;
  int m0 = (wave / ntile) << 4, n0 = (wave % ntile) << 4;
  int half = lane >> 4, l16 = lane & 15;
  const unsigned int* A32 = (const unsigned int*)A;   // 2 bf16 per dword
  const unsigned int* W32 = (const unsigned int*)W;
  int Ku = K >> 1;                                    // dwords per row
  int arow = (m0 + l16) * Ku + half * 8;              // A: half covers K 0-15 / 16-31
  int wrow = (n0 + l16) * Ku + half * 8;              // B: half covers K 0-15 / 16-31
  v8f c = {};
  for (int k0 = 0; k0 < Ku; k0 += 16) {               // 32 bf16 of K per iter
    AB a, b;
#pragma unroll
    for (int v = 0; v < 4; ++v) {
      a.u[v]     = A32[arow + k0 + v];
      a.u[v + 4] = A32[arow + k0 + 4 + v];
    }
#pragma unroll
    for (int v = 0; v < 8; ++v) b.u[v] = W32[wrow + k0 + v];
    c = __builtin_amdgcn_wmma_f32_16x16x32_bf16(false, a.v, false, b.v, (short)0, c, false, false);
  }
  int col = n0 + l16;
#pragma unroll
  for (int r = 0; r < 8; ++r) {
    int row = m0 + r + half * 8;                      // C layout: vgpr r -> rows r / r+8
    float val = c[r] * scale;
    if (outF) outF[(size_t)row * N + col] = val;
    if (outB) outB[(size_t)row * N + col] = f2bf(val);
  }
}

// ---------- per-row dots with att_src / att_dst (one wave per row) ----------
__global__ void k_rowdot(const float* __restrict__ x, const float* __restrict__ asv,
                         const float* __restrict__ adv, float* __restrict__ a_s,
                         float* __restrict__ a_d, int L, int D) {
  int wid = threadIdx.x >> 5, lane = threadIdx.x & 31;
  int row = blockIdx.x * (blockDim.x >> 5) + wid;
  if (row >= L) return;
  float s1 = 0.f, s2 = 0.f;
  for (int c = lane; c < D; c += 32) {
    float xv = x[(size_t)row * D + c];
    s1 += xv * asv[c];
    s2 += xv * adv[c];
  }
  for (int o = 16; o > 0; o >>= 1) {
    s1 += __shfl_xor(s1, o, 32);
    s2 += __shfl_xor(s2, o, 32);
  }
  if (lane == 0) { a_s[row] = s1; a_d[row] = s2; }
}

// ---------- edge softmax passes (self-loops appended as edges [E, E+L)) ----------
__global__ void k_edge_max(const int* __restrict__ ei, const float* __restrict__ a_s,
                           const float* __restrict__ a_d, float* __restrict__ e_buf,
                           float* __restrict__ mbuf, int E, int L) {
  int e = blockIdx.x * blockDim.x + threadIdx.x;
  int Et = E + L;
  if (e >= Et) return;
  int s, d;
  if (e < E) { s = ei[e]; d = ei[E + e]; } else { s = d = e - E; }
  float v = a_s[s] + a_d[d];
  v = (v > 0.f) ? v : LEAK * v;
  e_buf[e] = v;
  atomicMaxF(&mbuf[d], v);
}

__global__ void k_edge_exp(const int* __restrict__ ei, float* __restrict__ e_buf,
                           const float* __restrict__ mbuf, float* __restrict__ denom,
                           int E, int L) {
  int e = blockIdx.x * blockDim.x + threadIdx.x;
  int Et = E + L;
  if (e >= Et) return;
  int d = (e < E) ? ei[E + e] : (e - E);
  float ex = __expf(e_buf[e] - mbuf[d]);
  e_buf[e] = ex;
  atomicAdd(&denom[d], ex);
}

// 4 edges per 256-thread block, 4 columns per thread
__global__ void k_edge_scatter(const int* __restrict__ ei, const float* __restrict__ ex,
                               const float* __restrict__ x, float* __restrict__ acc,
                               int E, int L, int D) {
  int eloc = threadIdx.x >> 6;
  int cpt = D >> 6;
  int c0 = (threadIdx.x & 63) * cpt;
  int e = blockIdx.x * 4 + eloc;
  int Et = E + L;
  if (e >= Et) return;
  int s, d;
  if (e < E) { s = ei[e]; d = ei[E + e]; } else { s = d = e - E; }
  float w = ex[e];
  const float* xs = x + (size_t)s * D + c0;
  float* ac = acc + (size_t)d * D + c0;
  for (int i = 0; i < cpt; ++i) atomicAdd(&ac[i], w * xs[i]);
}

// h_struct = h + acc/denom + bias  -> bf16 (feeds K GEMM)
__global__ void k_hstruct(const float* __restrict__ h, const float* __restrict__ acc,
                          const float* __restrict__ denom, const float* __restrict__ bias,
                          unsigned short* __restrict__ hsb, int L, int D) {
  int i = blockIdx.x * blockDim.x + threadIdx.x;
  if (i >= L * D) return;
  int row = i / D, c = i - row * D;
  hsb[i] = f2bf(h[i] + acc[i] / denom[row] + bias[c]);
}

// ---------- two-pass flash-style column-mean of softmax(Q K^T) ----------
// One block = 4 waves sharing one 16-row Q tile (kept in registers per wave);
// wave w covers columns [w*L/4, (w+1)*L/4). Softmax (m,s) state merged via LDS.
// D must be 256.
__global__ void k_attn_scores(const unsigned short* __restrict__ Qb,
                              const unsigned short* __restrict__ Kb,
                              float* __restrict__ scores, int L, int D) {
  __shared__ float smrg[NSPLIT][16][2];
  int w = threadIdx.x >> 5;              // wave in block: column-split index
  int lane = threadIdx.x & 31;
  int m0 = blockIdx.x << 4;              // row tile (same for all waves in block)
  if (m0 >= L) return;
  int half = lane >> 4, l16 = lane & 15;
  const unsigned int* Q32 = (const unsigned int*)Qb;
  const unsigned int* K32 = (const unsigned int*)Kb;
  int Ku = D >> 1;                       // 128 dwords per row
  int qrow = (m0 + l16) * Ku + half * 8;
  AB afr[8];                             // whole 16x256 Q tile in regs, reused 2*128 times
#pragma unroll
  for (int kk = 0; kk < 8; ++kk) {
#pragma unroll
    for (int v = 0; v < 4; ++v) {
      afr[kk].u[v]     = Q32[qrow + kk * 16 + v];
      afr[kk].u[v + 4] = Q32[qrow + kk * 16 + 4 + v];
    }
  }
  float mrow[8], srow[8];
#pragma unroll
  for (int r = 0; r < 8; ++r) { mrow[r] = -3.0e38f; srow[r] = 0.f; }
  int kbase = l16 * Ku + half * 8;
  int nbeg = w * (L / NSPLIT), nend = (w + 1) * (L / NSPLIT);

  // ---- pass 1: per-lane online (max, sumexp) over this lane's column stream ----
  for (int n0 = nbeg; n0 < nend; n0 += 16) {
    if (n0 + 16 < nend) __builtin_prefetch(&K32[(n0 + 16) * Ku + kbase], 0, 1);
    v8f c = {};
#pragma unroll
    for (int kk = 0; kk < 8; ++kk) {
      AB b;
#pragma unroll
      for (int v = 0; v < 8; ++v) b.u[v] = K32[n0 * Ku + kbase + kk * 16 + v];
      c = __builtin_amdgcn_wmma_f32_16x16x32_bf16(false, afr[kk].v, false, b.v, (short)0, c, false, false);
    }
#pragma unroll
    for (int r = 0; r < 8; ++r) {
      float val = c[r];
      if (val <= mrow[r]) {
        srow[r] += __expf(val - mrow[r]);
      } else {
        srow[r] = srow[r] * __expf(mrow[r] - val) + 1.f;
        mrow[r] = val;
      }
    }
  }
  // merge (m,s) across the 16 lanes covering each row (intra-wave)
#pragma unroll
  for (int o = 1; o < 16; o <<= 1) {
#pragma unroll
    for (int r = 0; r < 8; ++r) {
      float om = __shfl_xor(mrow[r], o, 32);
      float os = __shfl_xor(srow[r], o, 32);
      float nm = fmaxf(mrow[r], om);
      srow[r] = srow[r] * __expf(mrow[r] - nm) + os * __expf(om - nm);
      mrow[r] = nm;
    }
  }
  // merge across the NSPLIT column-waves via LDS
  if (l16 == 0) {                        // lanes 0 and 16 publish rows 0-7 / 8-15
#pragma unroll
    for (int r = 0; r < 8; ++r) {
      smrg[w][r + half * 8][0] = mrow[r];
      smrg[w][r + half * 8][1] = srow[r];
    }
  }
  __syncthreads();
#pragma unroll
  for (int ow = 0; ow < NSPLIT; ++ow) {
    if (ow == w) continue;
#pragma unroll
    for (int r = 0; r < 8; ++r) {
      float om = smrg[ow][r + half * 8][0];
      float os = smrg[ow][r + half * 8][1];
      float nm = fmaxf(mrow[r], om);
      srow[r] = srow[r] * __expf(mrow[r] - nm) + os * __expf(om - nm);
      mrow[r] = nm;
    }
  }
#pragma unroll
  for (int r = 0; r < 8; ++r) srow[r] = 1.f / srow[r];   // inv denom

  // ---- pass 2: accumulate normalized probs into column sums ----
  for (int n0 = nbeg; n0 < nend; n0 += 16) {
    if (n0 + 16 < nend) __builtin_prefetch(&K32[(n0 + 16) * Ku + kbase], 0, 1);
    v8f c = {};
#pragma unroll
    for (int kk = 0; kk < 8; ++kk) {
      AB b;
#pragma unroll
      for (int v = 0; v < 8; ++v) b.u[v] = K32[n0 * Ku + kbase + kk * 16 + v];
      c = __builtin_amdgcn_wmma_f32_16x16x32_bf16(false, afr[kk].v, false, b.v, (short)0, c, false, false);
    }
    float t = 0.f;
#pragma unroll
    for (int r = 0; r < 8; ++r) t += __expf(c[r] - mrow[r]) * srow[r];
    t += __shfl_xor(t, 16, 32);            // fold the two half-wave row groups
    if (half == 0) atomicAdd(&scores[n0 + l16], t);
  }
}

// ---------- single-block LDS bitonic argsort (descending, ties by index) ----------
#define SORT_MAX 8192
__global__ void k_sort(const float* __restrict__ scores, float invL,
                       float* __restrict__ outPerm, float* __restrict__ outScores,
                       int* __restrict__ permI, int L, int Lp) {
  __shared__ float skey[SORT_MAX];
  __shared__ int   sidx[SORT_MAX];
  int tid = threadIdx.x, nt = blockDim.x;
  for (int i = tid; i < Lp; i += nt) {
    skey[i] = (i < L) ? scores[i] : -3.4e38f;
    sidx[i] = i;
  }
  __syncthreads();
  for (int k = 2; k <= Lp; k <<= 1) {
    for (int j = k >> 1; j > 0; j >>= 1) {
      for (int i = tid; i < Lp; i += nt) {
        int ixj = i ^ j;
        if (ixj > i) {
          float k1 = skey[i], k2 = skey[ixj];
          int i1 = sidx[i], i2 = sidx[ixj];
          bool before = (k1 > k2) || (k1 == k2 && i1 < i2);  // descending total order
          if (((i & k) == 0) != before) {
            skey[i] = k2; skey[ixj] = k1;
            sidx[i] = i2; sidx[ixj] = i1;
          }
        }
      }
      __syncthreads();
    }
  }
  for (int i = tid; i < L; i += nt) {
    outPerm[i] = (float)sidx[i];
    permI[i]   = sidx[i];
    outScores[i] = scores[i] * invL;       // mean over rows, original order
  }
}

__global__ void k_gather(const float* __restrict__ h, const int* __restrict__ permI,
                         float* __restrict__ out, int L, int D) {
  int row = blockIdx.x, c = threadIdx.x;
  if (row < L && c < D) out[(size_t)row * D + c] = h[(size_t)permI[row] * D + c];
}

// ---------------------------------------------------------------------------
extern "C" void kernel_launch(void* const* d_in, const int* in_sizes, int n_in,
                              void* d_out, int out_size, void* d_ws, size_t ws_size,
                              hipStream_t stream) {
  (void)n_in; (void)out_size; (void)ws_size;
  const float* h       = (const float*)d_in[0];
  const int*   ei      = (const int*)d_in[1];
  const float* W_Q     = (const float*)d_in[2];
  const float* W_K     = (const float*)d_in[3];
  const float* W_gat   = (const float*)d_in[4];
  const float* att_src = (const float*)d_in[5];
  const float* att_dst = (const float*)d_in[6];
  const float* bias    = (const float*)d_in[7];

  int D = in_sizes[5];
  int L = in_sizes[0] / D;
  int E = in_sizes[1] / 2;
  int Et = E + L;
  float qscale = 1.0f / sqrtf((float)D);

  char* ws = (char*)d_ws;
  size_t off = 0;
  auto alloc = [&](size_t bytes) -> char* {
    off = (off + 255) & ~(size_t)255;
    char* p = ws + off;
    off += bytes;
    return p;
  };
  unsigned short* h_bf  = (unsigned short*)alloc((size_t)L * D * 2);
  unsigned short* wg_bf = (unsigned short*)alloc((size_t)D * D * 2);
  unsigned short* wq_bf = (unsigned short*)alloc((size_t)D * D * 2);
  unsigned short* wk_bf = (unsigned short*)alloc((size_t)D * D * 2);
  unsigned short* hs_bf = (unsigned short*)alloc((size_t)L * D * 2);
  unsigned short* q_bf  = (unsigned short*)alloc((size_t)L * D * 2);
  unsigned short* k_bf  = (unsigned short*)alloc((size_t)L * D * 2);
  float* x     = (float*)alloc((size_t)L * D * 4);
  float* acc   = (float*)alloc((size_t)L * D * 4);
  float* a_s   = (float*)alloc((size_t)L * 4);
  float* a_d   = (float*)alloc((size_t)L * 4);
  float* mbuf  = (float*)alloc((size_t)L * 4);
  float* denom = (float*)alloc((size_t)L * 4);
  float* sc    = (float*)alloc((size_t)L * 4);
  float* e_buf = (float*)alloc((size_t)Et * 4);
  int*   permI = (int*)alloc((size_t)L * 4);

  float* outH      = (float*)d_out;
  float* outPerm   = outH + (size_t)L * D;
  float* outScores = outPerm + L;

  const int B = 256;
  // fp32 -> bf16 operand conversions
  k_cvt_bf16<<<(L * D + B - 1) / B, B, 0, stream>>>(h, h_bf, 1.f, L * D);
  k_cvt_bf16<<<(D * D + B - 1) / B, B, 0, stream>>>(W_gat, wg_bf, 1.f, D * D);
  k_cvt_bf16<<<(D * D + B - 1) / B, B, 0, stream>>>(W_Q, wq_bf, 1.f, D * D);
  k_cvt_bf16<<<(D * D + B - 1) / B, B, 0, stream>>>(W_K, wk_bf, 1.f, D * D);
  // init accumulators
  (void)hipMemsetAsync(acc, 0, (size_t)L * D * 4, stream);
  (void)hipMemsetAsync(denom, 0, (size_t)L * 4, stream);
  (void)hipMemsetAsync(sc, 0, (size_t)L * 4, stream);
  k_fill<<<(L + B - 1) / B, B, 0, stream>>>(mbuf, -3.0e38f, L);

  int gw = (L / 16) * (D / 16);   // waves per LxD GEMM
  // x = h @ W_gat^T  (fp32 out: reused by edge scatter)
  k_gemm_bf16<<<(gw * 32 + B - 1) / B, B, 0, stream>>>(h_bf, wg_bf, x, nullptr, 1.f, L, D, D);
  // attention logits per node
  k_rowdot<<<(L + 7) / 8, B, 0, stream>>>(x, att_src, att_dst, a_s, a_d, L, D);
  // edge softmax (max, exp/denom, weighted scatter)
  k_edge_max<<<(Et + B - 1) / B, B, 0, stream>>>(ei, a_s, a_d, e_buf, mbuf, E, L);
  k_edge_exp<<<(Et + B - 1) / B, B, 0, stream>>>(ei, e_buf, mbuf, denom, E, L);
  k_edge_scatter<<<(Et + 3) / 4, B, 0, stream>>>(ei, e_buf, x, acc, E, L, D);
  // h_struct (bf16)
  k_hstruct<<<(L * D + B - 1) / B, B, 0, stream>>>(h, acc, denom, bias, hs_bf, L, D);
  // Q (pre-scaled by 1/sqrt(D)) and K, straight to bf16
  k_gemm_bf16<<<(gw * 32 + B - 1) / B, B, 0, stream>>>(h_bf, wq_bf, nullptr, q_bf, qscale, L, D, D);
  k_gemm_bf16<<<(gw * 32 + B - 1) / B, B, 0, stream>>>(hs_bf, wk_bf, nullptr, k_bf, 1.f, L, D, D);
  // column means of softmax(QK^T): one block per 16-row tile, 4 column-split waves
  k_attn_scores<<<L / 16, 32 * NSPLIT, 0, stream>>>(q_bf, k_bf, sc, L, D);
  // argsort + outputs
  int Lp = 1; while (Lp < L) Lp <<= 1;
  k_sort<<<1, 1024, 0, stream>>>(sc, 1.0f / (float)L, outPerm, outScores, permI, L, Lp);
  k_gather<<<L, D, 0, stream>>>(h, permI, outH, L, D);
}
